// MultiheadLMAttentionWithCache_21912923144409
// MI455X (gfx1250) — matrix operand order repaired and checked by reading, not compiled, not run on hardware
//
#include <hip/hip_runtime.h>
#include <stdint.h>

// ---------------------------------------------------------------------------
// MI455X (gfx1250) multi-head causal attention w/ KV cache, bf16 WMMA path.
// Shapes (from reference): T=1024 B=4 E=2048 H=16 hd=128 MAX_LEN=4096
// ---------------------------------------------------------------------------

typedef __bf16 bf16;
typedef __attribute__((ext_vector_type(16))) __bf16 v16bf;
typedef __attribute__((ext_vector_type(8)))  __bf16 v8bf;
typedef __attribute__((ext_vector_type(8)))  float  v8f;

static __device__ __forceinline__ v16bf cat8(v8bf lo, v8bf hi) {
  return __builtin_shufflevector(lo, hi, 0,1,2,3,4,5,6,7,8,9,10,11,12,13,14,15);
}
static __device__ __forceinline__ v8f wmma_bf16(v16bf a, v16bf b, v8f c) {
  // (neg_a, A, neg_b, B, c_mod, C, reuse_a, reuse_b)
  return __builtin_amdgcn_wmma_f32_16x16x32_bf16(false, a, false, b, (short)0, c, false, false);
}
static __device__ __forceinline__ float hmax16(float v) {
  v = fmaxf(v, __shfl_xor(v, 1, 32));
  v = fmaxf(v, __shfl_xor(v, 2, 32));
  v = fmaxf(v, __shfl_xor(v, 4, 32));
  v = fmaxf(v, __shfl_xor(v, 8, 32));
  return v;
}
static __device__ __forceinline__ float hsum16(float v) {
  v += __shfl_xor(v, 1, 32);
  v += __shfl_xor(v, 2, 32);
  v += __shfl_xor(v, 4, 32);
  v += __shfl_xor(v, 8, 32);
  return v;
}
// gfx1250 async global->LDS copy (16B per lane), tracked by ASYNCcnt.
// lds_off: 32-bit LDS byte offset (low 32 bits of generic pointer to LDS).
static __device__ __forceinline__ void async_load_b128(uint32_t lds_off, const void* gptr) {
  asm volatile("global_load_async_to_lds_b128 %0, %1, off"
               :: "v"(lds_off), "v"((uint64_t)(uintptr_t)gptr)
               : "memory");
}
static __device__ __forceinline__ void wait_async0() {
  asm volatile("s_wait_asynccnt 0" ::: "memory");
}

// ===========================================================================
// Kernel 1: qkv = x @ w_in^T + b_in ; scatter q(scaled)/k/v -> bf16 workspace
//   A = x       (4096 x 2048) f32 row-major   (row m = t*4+b)
//   B = w_in    (6144 x 2048) f32 row-major   (out = A @ B^T)
// Block: 64x64 tile, 128 threads = 4 waves (2x2 of 32x32 wave tiles).
// ===========================================================================
__global__ __launch_bounds__(128)
void qkv_kernel(const float* __restrict__ x, const float* __restrict__ w_in,
                const float* __restrict__ b_in,
                bf16* __restrict__ q_ws, bf16* __restrict__ k_ws,
                bf16* __restrict__ v_ws) {
  __shared__ bf16 As[64][40];   // 32 K + 8 pad (row stride 80B, 16B multiple)
  __shared__ bf16 Bs[64][40];

  const int tid  = threadIdx.x;
  const int lane = tid & 31;
  const int w    = tid >> 5;
  const int m0   = blockIdx.y * 64;
  const int n0   = blockIdx.x * 64;
  const int wm   = (w >> 1) * 32;
  const int wn   = (w & 1) * 32;
  const int fr   = lane & 15;          // fragment row / column
  const int kbA  = (lane >> 4) * 8;    // A-fragment K base (split 8+8)
  const int kbB  = (lane >> 4) * 16;   // unused; B fragments also load with kbA (split 8+8)
  const int lrow = tid >> 1;           // cooperative-fill row 0..63
  const int lcol = (tid & 1) * 16;     // cooperative-fill col half

  v8f acc[2][2] = {};

  for (int k0 = 0; k0 < 2048; k0 += 32) {
    __syncthreads();
    { // fill LDS tiles (f32 -> bf16 convert on the fly)
      const float4* ax = (const float4*)(x    + (size_t)(m0 + lrow) * 2048 + k0 + lcol);
      const float4* bx = (const float4*)(w_in + (size_t)(n0 + lrow) * 2048 + k0 + lcol);
#pragma unroll
      for (int i = 0; i < 4; ++i) {
        float4 va = ax[i], vb = bx[i];
        As[lrow][lcol + i*4 + 0] = (bf16)va.x; As[lrow][lcol + i*4 + 1] = (bf16)va.y;
        As[lrow][lcol + i*4 + 2] = (bf16)va.z; As[lrow][lcol + i*4 + 3] = (bf16)va.w;
        Bs[lrow][lcol + i*4 + 0] = (bf16)vb.x; Bs[lrow][lcol + i*4 + 1] = (bf16)vb.y;
        Bs[lrow][lcol + i*4 + 2] = (bf16)vb.z; Bs[lrow][lcol + i*4 + 3] = (bf16)vb.w;
      }
    }
    __syncthreads();

    v16bf a[2], b[2];
#pragma unroll
    for (int i = 0; i < 2; ++i) {
      const bf16* p = &As[wm + i*16 + fr][0];
      a[i] = cat8(*(const v8bf*)(p + kbA), *(const v8bf*)(p + kbA + 16));
    }
#pragma unroll
    for (int j = 0; j < 2; ++j) {
      const bf16* p = &Bs[wn + j*16 + fr][0];
      b[j] = cat8(*(const v8bf*)(p + kbA), *(const v8bf*)(p + kbA + 16));
    }
#pragma unroll
    for (int i = 0; i < 2; ++i)
#pragma unroll
      for (int j = 0; j < 2; ++j)
        acc[i][j] = wmma_bf16(a[i], b[j], acc[i][j]);
  }

  // epilogue: +bias, split qkv per reshape(T, B*H, 3*hd), scale q
  const float scaling = 0.08838834764831845f;  // 128^-0.5
#pragma unroll
  for (int i = 0; i < 2; ++i)
#pragma unroll
    for (int j = 0; j < 2; ++j)
#pragma unroll
      for (int r = 0; r < 8; ++r) {
        const int ml = wm + i*16 + (lane >> 4) * 8 + r;
        const int nl = wn + j*16 + fr;
        const int m  = m0 + ml;          // row = t*4 + b
        const int c  = n0 + nl;          // out channel in [0, 6144)
        float val    = acc[i][j][r] + b_in[c];
        const int h  = c / 384;
        const int rr = c - h * 384;      // within-head: [q(128) k(128) v(128)]
        const int t  = m >> 2;
        const int bb = m & 3;
        const size_t head = (size_t)(bb * 16 + h);
        if (rr < 128)
          q_ws[(head * 1024 + t) * 128 + rr]       = (bf16)(val * scaling);
        else if (rr < 256)
          k_ws[(head * 1024 + t) * 128 + rr - 128] = (bf16)val;
        else
          v_ws[(head * 1024 + t) * 128 + rr - 256] = (bf16)val;
      }
}

// ===========================================================================
// Kernel 2: flash attention per (head n, 64-query tile).
// Context rows s < start come from f32 caches (VGPR path w/ cvt), rows >=
// start come from bf16 workspace via async global->LDS DMA.
// ===========================================================================
__global__ __launch_bounds__(128)
void attn_kernel(const float* __restrict__ k_cache, const float* __restrict__ v_cache,
                 const bf16* __restrict__ q_ws, const bf16* __restrict__ k_ws,
                 const bf16* __restrict__ v_ws, bf16* __restrict__ attn_ws,
                 const int* __restrict__ p_input_id, const int* __restrict__ p_slice_len) {
  __shared__ bf16 Kt[64][136];     // [s][hd],  row stride 272B
  __shared__ bf16 Vt[128][72];     // [hd][s],  transposed, row stride 144B
  __shared__ bf16 Pex[4][16][72];  // per-wave probs strip (D-layout -> A-layout)

  const int tid  = threadIdx.x;
  const int lane = tid & 31;
  const int w    = tid >> 5;
  const int n    = blockIdx.y;             // head 0..63
  const int t0   = blockIdx.x * 64;
  const int start = p_input_id[0] * p_slice_len[0];   // 3072
  const int fr   = lane & 15;
  const int kbA  = (lane >> 4) * 8;
  const int kbB  = (lane >> 4) * 16;
  const int frow = tid >> 1;               // cooperative-fill s-row 0..63
  const int fc   = (tid & 1) * 64;         // cooperative-fill hd half

  // Q strip (16 rows x 128) in A-fragments, stays in registers
  v16bf qf[4];
  {
    const bf16* qrow = q_ws + ((size_t)n * 1024 + t0 + w * 16 + fr) * 128;
#pragma unroll
    for (int kc = 0; kc < 4; ++kc)
      qf[kc] = cat8(*(const v8bf*)(qrow + kc*32 + kbA),
                    *(const v8bf*)(qrow + kc*32 + kbA + 16));
  }

  v8f o[8] = {};
  float mi[8], li[8];
#pragma unroll
  for (int r = 0; r < 8; ++r) { mi[r] = -1e38f; li[r] = 0.f; }

  const int ntiles = (start + t0 + 63) / 64 + 1;
  for (int st = 0; st < ntiles; ++st) {
    const int s0 = st * 64;
    { // ---- fill K tile + transposed V tile ----
      const int sg = s0 + frow;
      if (sg < start) {
        const float4* ks = (const float4*)(k_cache + ((size_t)n * 4096 + sg) * 128 + fc);
        const float4* vs = (const float4*)(v_cache + ((size_t)n * 4096 + sg) * 128 + fc);
#pragma unroll
        for (int i = 0; i < 16; ++i) {
          float4 kv = ks[i], vv = vs[i];
          Kt[frow][fc + i*4 + 0] = (bf16)kv.x; Kt[frow][fc + i*4 + 1] = (bf16)kv.y;
          Kt[frow][fc + i*4 + 2] = (bf16)kv.z; Kt[frow][fc + i*4 + 3] = (bf16)kv.w;
          Vt[fc + i*4 + 0][frow] = (bf16)vv.x; Vt[fc + i*4 + 1][frow] = (bf16)vv.y;
          Vt[fc + i*4 + 2][frow] = (bf16)vv.z; Vt[fc + i*4 + 3][frow] = (bf16)vv.w;
        }
      } else {
        // K rows already bf16 in workspace: async DMA straight into LDS.
        const bf16* ksrc = k_ws + ((size_t)n * 1024 + (sg - start)) * 128 + fc;
        const uint32_t koff = (uint32_t)(uintptr_t)&Kt[frow][fc];
#pragma unroll
        for (int i = 0; i < 8; ++i)
          async_load_b128(koff + i * 16, ksrc + i * 8);
        // V needs the [hd][s] transpose: VGPR path with scalar LDS scatter.
        const bf16* vs = v_ws + ((size_t)n * 1024 + (sg - start)) * 128 + fc;
#pragma unroll
        for (int i = 0; i < 64; ++i) Vt[fc + i][frow] = vs[i];
      }
      // prefetch next tile's cache lines (global_prefetch_b8)
      if (st + 1 < ntiles && s0 + 64 + frow < start) {
        __builtin_prefetch(k_cache + ((size_t)n * 4096 + s0 + 64 + frow) * 128 + fc, 0, 0);
        __builtin_prefetch(v_cache + ((size_t)n * 4096 + s0 + 64 + frow) * 128 + fc, 0, 0);
      }
    }
    wait_async0();        // async LDS writes must land before the barrier
    __syncthreads();

    // ---- scores S = Q @ K^T (per-wave 16x64 strip) ----
    v8f sc[4];
#pragma unroll
    for (int j = 0; j < 4; ++j) {
      v8f z = {};
      sc[j] = z;
#pragma unroll
      for (int kc = 0; kc < 4; ++kc) {
        const bf16* kr = &Kt[j*16 + fr][kc*32];
        v16bf bfr = cat8(*(const v8bf*)(kr + kbA), *(const v8bf*)(kr + kbA + 16));
        sc[j] = wmma_bf16(qf[kc], bfr, sc[j]);
      }
    }
    // ---- causal mask (boundary tile only) ----
    if (s0 + 63 > start + t0) {
#pragma unroll
      for (int j = 0; j < 4; ++j) {
        const int s = s0 + j*16 + fr;
#pragma unroll
        for (int r = 0; r < 8; ++r) {
          const int t = t0 + w*16 + (lane >> 4) * 8 + r;
          if (s > start + t) sc[j][r] = -1e30f;
        }
      }
    }
    // ---- online softmax (rows live in 16-lane half-groups) ----
#pragma unroll
    for (int r = 0; r < 8; ++r) {
      float mx = fmaxf(fmaxf(sc[0][r], sc[1][r]), fmaxf(sc[2][r], sc[3][r]));
      mx = hmax16(mx);
      const float mnew = fmaxf(mi[r], mx);
      const float corr = __expf(mi[r] - mnew);
      mi[r] = mnew;
      float ts = 0.f;
#pragma unroll
      for (int j = 0; j < 4; ++j) {
        float p = __expf(sc[j][r] - mnew);
        sc[j][r] = p;
        ts += p;
      }
      li[r] = li[r] * corr + hsum16(ts);
#pragma unroll
      for (int j2 = 0; j2 < 8; ++j2) o[j2][r] *= corr;
    }
    // ---- probs: D-layout -> LDS -> A-fragment layout ----
#pragma unroll
    for (int j = 0; j < 4; ++j)
#pragma unroll
      for (int r = 0; r < 8; ++r)
        Pex[w][(lane >> 4) * 8 + r][j*16 + fr] = (bf16)sc[j][r];
    __syncthreads();

    // ---- O += P @ V ----
#pragma unroll
    for (int kc = 0; kc < 2; ++kc) {
      const bf16* pr = &Pex[w][fr][kc*32];
      v16bf pa = cat8(*(const v8bf*)(pr + kbA), *(const v8bf*)(pr + kbA + 16));
#pragma unroll
      for (int j2 = 0; j2 < 8; ++j2) {
        const bf16* vr = &Vt[j2*16 + fr][kc*32];
        v16bf vb = cat8(*(const v8bf*)(vr + kbA), *(const v8bf*)(vr + kbA + 16));
        o[j2] = wmma_bf16(pa, vb, o[j2]);
      }
    }
    __syncthreads();
  }

  // ---- epilogue: normalize and store to (T, B, E) bf16 workspace ----
  const int bq = n >> 4, hq = n & 15;
#pragma unroll
  for (int r = 0; r < 8; ++r) {
    const float inv = 1.f / li[r];
    const int t = t0 + w*16 + (lane >> 4) * 8 + r;
    const size_t base = ((size_t)t * 4 + bq) * 2048 + hq * 128;
#pragma unroll
    for (int j2 = 0; j2 < 8; ++j2)
      attn_ws[base + j2*16 + fr] = (bf16)(o[j2][r] * inv);
  }
}

// ===========================================================================
// Kernel 3: out = attn @ w_out^T + b_out   (4096 x 2048) <- (4096x2048)(2048x2048)
// A tile is bf16 already: async global->LDS DMA. B tile converts f32->bf16.
// ===========================================================================
__global__ __launch_bounds__(128)
void outproj_kernel(const bf16* __restrict__ attn, const float* __restrict__ w_out,
                    const float* __restrict__ b_out, float* __restrict__ out) {
  __shared__ bf16 As[64][40];
  __shared__ bf16 Bs[64][40];

  const int tid  = threadIdx.x;
  const int lane = tid & 31;
  const int w    = tid >> 5;
  const int m0   = blockIdx.y * 64;
  const int n0   = blockIdx.x * 64;
  const int wm   = (w >> 1) * 32;
  const int wn   = (w & 1) * 32;
  const int fr   = lane & 15;
  const int kbA  = (lane >> 4) * 8;
  const int kbB  = (lane >> 4) * 16;
  const int lrow = tid >> 1;
  const int lcol = (tid & 1) * 16;

  v8f acc[2][2] = {};

  for (int k0 = 0; k0 < 2048; k0 += 32) {
    __syncthreads();
    {
      // A (bf16): async DMA, 2 x 16B per thread
      const bf16* ap = attn + (size_t)(m0 + lrow) * 2048 + k0 + lcol;
      const uint32_t aoff = (uint32_t)(uintptr_t)&As[lrow][lcol];
      async_load_b128(aoff,      ap);
      async_load_b128(aoff + 16, ap + 8);
      // B (f32 weights): VGPR path with packed cvt
      const float4* bx = (const float4*)(w_out + (size_t)(n0 + lrow) * 2048 + k0 + lcol);
#pragma unroll
      for (int i = 0; i < 4; ++i) {
        float4 vb = bx[i];
        Bs[lrow][lcol + i*4 + 0] = (bf16)vb.x; Bs[lrow][lcol + i*4 + 1] = (bf16)vb.y;
        Bs[lrow][lcol + i*4 + 2] = (bf16)vb.z; Bs[lrow][lcol + i*4 + 3] = (bf16)vb.w;
      }
    }
    wait_async0();
    __syncthreads();

    v16bf a[2], b[2];
#pragma unroll
    for (int i = 0; i < 2; ++i) {
      const bf16* p = &As[wm + i*16 + fr][0];
      a[i] = cat8(*(const v8bf*)(p + kbA), *(const v8bf*)(p + kbA + 16));
    }
#pragma unroll
    for (int j = 0; j < 2; ++j) {
      const bf16* p = &Bs[wn + j*16 + fr][0];
      b[j] = cat8(*(const v8bf*)(p + kbA), *(const v8bf*)(p + kbA + 16));
    }
#pragma unroll
    for (int i = 0; i < 2; ++i)
#pragma unroll
      for (int j = 0; j < 2; ++j)
        acc[i][j] = wmma_bf16(a[i], b[j], acc[i][j]);
  }

#pragma unroll
  for (int i = 0; i < 2; ++i)
#pragma unroll
    for (int j = 0; j < 2; ++j)
#pragma unroll
      for (int r = 0; r < 8; ++r) {
        const int m = m0 + wm + i*16 + (lane >> 4) * 8 + r;
        const int c = n0 + wn + j*16 + fr;
        out[(size_t)m * 2048 + c] = acc[i][j][r] + b_out[c];
      }
}

// ===========================================================================
extern "C" void kernel_launch(void* const* d_in, const int* in_sizes, int n_in,
                              void* d_out, int out_size, void* d_ws, size_t ws_size,
                              hipStream_t stream) {
  const float* x        = (const float*)d_in[0];
  const float* k_cache  = (const float*)d_in[1];
  const float* v_cache  = (const float*)d_in[2];
  const float* w_in     = (const float*)d_in[3];
  const float* b_in     = (const float*)d_in[4];
  const float* w_out    = (const float*)d_in[5];
  const float* b_out    = (const float*)d_in[6];
  const int*   input_id = (const int*)d_in[7];
  const int*   slice_len= (const int*)d_in[8];
  float* out = (float*)d_out;

  // workspace: 4 x 16MB bf16 intermediates (q, fresh-k, fresh-v, attn)
  bf16* q_ws    = (bf16*)d_ws;
  bf16* k_ws    = q_ws + (size_t)64 * 1024 * 128;
  bf16* v_ws    = k_ws + (size_t)64 * 1024 * 128;
  bf16* attn_ws = v_ws + (size_t)64 * 1024 * 128;

  qkv_kernel<<<dim3(96, 64), 128, 0, stream>>>(x, w_in, b_in, q_ws, k_ws, v_ws);
  attn_kernel<<<dim3(16, 64), 128, 0, stream>>>(k_cache, v_cache, q_ws, k_ws, v_ws,
                                                attn_ws, input_id, slice_len);
  outproj_kernel<<<dim3(32, 64), 128, 0, stream>>>(attn_ws, w_out, b_out, out);
}
